// LSTMModel_7155415515789
// MI455X (gfx1250) — matrix operand order repaired and checked by reading, not compiled, not run on hardware
//
#include <hip/hip_runtime.h>
#include <cstdint>
#include <cstddef>

// Problem constants (match reference)
#define B_ 64
#define T_ 512
#define I_ 512
#define H_ 1024
#define O_ 512

typedef __attribute__((ext_vector_type(16))) __bf16 v16bf;
typedef __attribute__((ext_vector_type(8)))  float  v8f;
typedef __attribute__((ext_vector_type(4)))  unsigned int v4u;
typedef __attribute__((ext_vector_type(8)))  int    v8i;
typedef __attribute__((ext_vector_type(4)))  int    v4i;

union AFrag { uint4 q[2]; v16bf v; };

// LDS staging geometry for the recurrent layer: 16 rows x 1024 bf16, with TDM
// padding of 4 DWORDs (8 elements) every 256 DWORDs (512 elements) so row
// stride = 1040 elements (520 DWORDs -> lanes land 8 banks apart, 2-way max).
#define LDS_ROW_ELEMS 1040
#define LDS_TILE_Q    ((16 * LDS_ROW_ELEMS) / 8)   // uint4 count (33,280 B)

__device__ __forceinline__ unsigned short f2bf(float f) {
    unsigned int u = __float_as_uint(f);
    u += 0x7FFFu + ((u >> 16) & 1u);   // round-to-nearest-even
    return (unsigned short)(u >> 16);
}

// ---------------------------------------------------------------------------
// Pack a row-major fp32 weight [N, K] (out-major, torch W[out,in]) into WMMA
// B-operand tile order for v_wmma_f32_16x16x32_bf16:
//   tile (ntile, kc): 512 bf16; element (lane, e) = W[ntile*16 + lane%16,
//                     kc*32 + (lane>=16?8:0) + (e&7) + (e>=8?16:0)]
// ---------------------------------------------------------------------------
__global__ void pack_b_kernel(const float* __restrict__ W,
                              unsigned short* __restrict__ P,
                              int KC, int K, int total) {
    int idx = blockIdx.x * 256 + threadIdx.x;
    if (idx >= total) return;
    int e    = idx & 15;
    int lane = (idx >> 4) & 31;
    int tk   = idx >> 9;
    int kc   = tk % KC;
    int n    = tk / KC;
    int col  = n * 16 + (lane & 15);
    int k    = kc * 32 + ((lane & 16) ? 8 : 0) + (e & 7) + ((e & 8) ? 16 : 0);
    P[idx] = f2bf(W[(size_t)col * K + k]);
}

__global__ void cast_bf16_kernel(const float* __restrict__ in,
                                 unsigned short* __restrict__ out, int n) {
    int i = blockIdx.x * 256 + threadIdx.x;
    if (i < n) out[i] = f2bf(in[i]);
}

__global__ void vec_add_kernel(const float* __restrict__ a,
                               const float* __restrict__ b,
                               float* __restrict__ o, int n) {
    int i = blockIdx.x * 256 + threadIdx.x;
    if (i < n) o[i] = a[i] + b[i];
}

// ---------------------------------------------------------------------------
// Generic bf16 WMMA GEMM: Out[M,N] = A[M,K](bf16,row-major) * Bpacked + bias.
// Block: 256 thr = 8 waves; block tile 128(M) x 64(N); wave tile 64 x 16
// (4 accumulators, 1 B-fragment reused 4x per K-chunk of 32).
// remapB>0: input row r = t*remapB + b  ->  output row b*remapT + t (for FC).
// ---------------------------------------------------------------------------
__global__ __launch_bounds__(256) void gemm_bf16_wmma(
    const unsigned short* __restrict__ A,
    const unsigned short* __restrict__ Bp,
    const float* __restrict__ bias,
    float* __restrict__ Out,
    int M, int N, int K, int remapB, int remapT)
{
    const int lane = threadIdx.x & 31;
    const int wave = threadIdx.x >> 5;
    const int mg   = wave >> 2;                  // 0..1
    const int ng   = wave & 3;                   // 0..3
    const int rowBase = blockIdx.x * 128 + mg * 64;
    const int ntile   = blockIdx.y * 4 + ng;
    const int KC      = K >> 5;
    const int kb_lane = (lane & 16) ? 8 : 0;     // A-operand K sub-offset per lane half
    const int lr      = lane & 15;

    v8f acc[4] = {};
    const unsigned short* bptr = Bp + (size_t)ntile * KC * 512 + lane * 16;
    for (int kc = 0; kc < KC; ++kc) {
        v16bf bf = *reinterpret_cast<const v16bf*>(bptr);
        __builtin_prefetch((const void*)(bptr + 512), 0, 3);   // global_prefetch_b8 (all levels)
        bptr += 512;
#pragma unroll
        for (int s = 0; s < 4; ++s) {
            const unsigned short* arow =
                A + (size_t)(rowBase + s * 16 + lr) * K + kc * 32 + kb_lane;
            AFrag a;
            a.q[0] = *reinterpret_cast<const uint4*>(arow);
            a.q[1] = *reinterpret_cast<const uint4*>(arow + 16);
            acc[s] = __builtin_amdgcn_wmma_f32_16x16x32_bf16(
                false, a.v, false, bf, (short)0, acc[s], false, false);
        }
    }

    const int col  = ntile * 16 + lr;
    const float bc = bias ? bias[col] : 0.f;
#pragma unroll
    for (int s = 0; s < 4; ++s) {
#pragma unroll
        for (int r = 0; r < 8; ++r) {
            int row = rowBase + s * 16 + ((lane & 16) ? (r + 8) : r);
            size_t orow;
            if (remapB > 0) {
                int t = row / remapB;
                int b = row - t * remapB;
                orow = (size_t)b * remapT + t;
            } else {
                orow = (size_t)row;
            }
            Out[orow * (size_t)N + col] = acc[s][r] + bc;
        }
    }
}

// ---------------------------------------------------------------------------
// Persistent LSTM layer. 32 blocks x 8 waves = 256 waves. Block b owns batch
// block rb = b&3 (16 rows, all its waves share them) and h-col tiles
// ct = (b>>2)*8 + wave. Cell state c lives in registers for all T steps.
//
// Per step:
//   1. wave 0 issues TENSOR_LOAD_TO_LDS (TDM): 16x1024 bf16 strip of h(t)
//      -> LDS, with pad_interval=256 DW / pad_amount=4 DW for bank spread;
//      waits s_wait_tensorcnt 0; __syncthreads publishes the tile.
//   2. 32 K-chunks x 4 gate WMMAs (A-fragments via ds_load_b128 from LDS,
//      B-fragments streamed from L2-resident packed W_hh).
//   3. fused sigmoid/tanh epilogue; h(t+1) stored bf16 to hbuf.
//   4. device-wide generation barrier (atomic counter + threadfence).
// ---------------------------------------------------------------------------
__global__ __launch_bounds__(256) void lstm_layer_persistent(
    unsigned short* __restrict__ hbuf,       // [(T+1)*B, H] bf16; block 0 zeroed
    const unsigned short* __restrict__ Whh,  // packed: 256 ntiles x 32 kc x 512
    const float* __restrict__ xproj,         // precomputed gate projections (fp32)
    long long xp_tstride,                    // floats per timestep
    long long xp_bstride,                    // floats per batch row
    unsigned* __restrict__ bar,              // zeroed generation counter
    int T)
{
    __shared__ uint4 smemq[LDS_TILE_Q];      // 16 x 1040 bf16 (TDM-padded)

    const int lane = threadIdx.x & 31;
    const int wave = threadIdx.x >> 5;
    const int rb   = blockIdx.x & 3;               // batch block (16 rows)
    const int ct   = (blockIdx.x >> 2) * 8 + wave; // h-col tile (0..63)
    const int kb_lane = (lane & 16) ? 8 : 0;
    const int col  = ct * 16 + (lane & 15);
    const unsigned lds_base = (unsigned)(uintptr_t)(&smemq[0]); // LDS byte offset

    float creg[8] = {0.f, 0.f, 0.f, 0.f, 0.f, 0.f, 0.f, 0.f};

    for (int t = 0; t < T; ++t) {
        // ---- 1. TDM: stage h(t) rows [rb*16, rb*16+16) x H into LDS ----
        if (wave == 0) {
            unsigned long long ga = (unsigned long long)(uintptr_t)hbuf
                                  + ((size_t)t * (B_ * H_) + (size_t)rb * 16 * H_) * 2ull;
            v4u g0;
            g0[0] = 1u;                                    // count=1 (valid user D#)
            g0[1] = lds_base;                              // LDS dest (bytes)
            g0[2] = (unsigned)(ga & 0xFFFFFFFFu);          // global_addr[31:0]
            g0[3] = (unsigned)((ga >> 32) & 0x01FFFFFFu)   // global_addr[56:32]
                  | (2u << 30);                            // type=2 ("image")
            v8i g1;
            g1[0] = (int)((1u << 16)                       // data_size=1 -> 2 bytes
                  | (1u << 20)                             // pad_enable
                  | (7u << 22)                             // pad_interval: 256 DWORDs
                  | (3u << 25));                           // pad_amount: 4 DWORDs
            g1[1] = (int)((H_ & 0xFFFFu) << 16);           // tensor_dim0[15:0]
            g1[2] = (int)((H_ >> 16) | (16u << 16));       // tensor_dim0 hi | tensor_dim1=16
            g1[3] = (int)((H_ & 0xFFFFu) << 16);           // tensor_dim1 hi=0 | tile_dim0=H
            g1[4] = 16;                                    // tile_dim1=16, tile_dim2=0
            g1[5] = (int)H_;                               // tensor_dim0_stride[31:0]
            g1[6] = 0;                                     // stride0 hi | stride1 lo
            g1[7] = 0;                                     // stride1 hi
            v4i gz4 = {0, 0, 0, 0};                        // groups 2/3 unused (2-D)
            v8i gz8 = {0, 0, 0, 0, 0, 0, 0, 0};
            __builtin_amdgcn_tensor_load_to_lds(g0, g1, gz4, gz4, gz8, 0);
            __builtin_amdgcn_s_wait_tensorcnt(0);
        }
        __syncthreads();

        // ---- 2. recurrent GEMM tile: 4 gate accumulators ----
        v8f acc[4] = {};
        const int lrow = (lane & 15) * LDS_ROW_ELEMS;
#pragma unroll 4
        for (int kc = 0; kc < (H_ / 32); ++kc) {
            const int k   = kc * 32 + kb_lane;
            const int off = lrow + k + ((k >> 9) << 3);    // +8 elems past pad point
            AFrag a;
            a.q[0] = smemq[off >> 3];
            a.q[1] = smemq[(off >> 3) + 2];
#pragma unroll
            for (int g = 0; g < 4; ++g) {
                const unsigned short* bptr =
                    Whh + (size_t)((g * (H_ / 16) + ct) * (H_ / 32) + kc) * 512 + lane * 16;
                v16bf bf = *reinterpret_cast<const v16bf*>(bptr);
                acc[g] = __builtin_amdgcn_wmma_f32_16x16x32_bf16(
                    false, a.v, false, bf, (short)0, acc[g], false, false);
            }
        }

        // ---- 3. fused LSTM cell epilogue ----
        const float* xpt = xproj + (size_t)t * xp_tstride;
        unsigned short* Hout = hbuf + (size_t)(t + 1) * (B_ * H_);
#pragma unroll
        for (int r = 0; r < 8; ++r) {
            int b = rb * 16 + ((lane & 16) ? (r + 8) : r);
            const float* xr = xpt + (size_t)b * xp_bstride;
            float gi = acc[0][r] + xr[0 * H_ + col];
            float gf = acc[1][r] + xr[1 * H_ + col];
            float gg = acc[2][r] + xr[2 * H_ + col];
            float go = acc[3][r] + xr[3 * H_ + col];
            gi = 1.f / (1.f + __expf(-gi));
            gf = 1.f / (1.f + __expf(-gf));
            gg = tanhf(gg);
            go = 1.f / (1.f + __expf(-go));
            float c = gf * creg[r] + gi * gg;
            creg[r] = c;
            Hout[(size_t)b * H_ + col] = f2bf(go * tanhf(c));
        }

        // ---- 4. device-wide step barrier (release/acquire) ----
        __threadfence();
        __syncthreads();
        if (threadIdx.x == 0) {
            atomicAdd(bar, 1u);
            unsigned target = (unsigned)(t + 1) * gridDim.x;
            volatile unsigned* vb = bar;
            while (*vb < target) { __builtin_amdgcn_s_sleep(1); }
        }
        __syncthreads();
        __threadfence();
    }
}

// ---------------------------------------------------------------------------
// Host-side orchestration (graph-capture safe: only kernel launches + memsets)
// ---------------------------------------------------------------------------
extern "C" void kernel_launch(void* const* d_in, const int* in_sizes, int n_in,
                              void* d_out, int out_size, void* d_ws, size_t ws_size,
                              hipStream_t stream) {
    const float* x    = (const float*)d_in[0];
    const float* Wih1 = (const float*)d_in[1];
    const float* Whh1 = (const float*)d_in[2];
    const float* bih1 = (const float*)d_in[3];
    const float* bhh1 = (const float*)d_in[4];
    const float* Wih2 = (const float*)d_in[5];
    const float* Whh2 = (const float*)d_in[6];
    const float* bih2 = (const float*)d_in[7];
    const float* bhh2 = (const float*)d_in[8];
    const float* fcW  = (const float*)d_in[9];
    const float* fcb  = (const float*)d_in[10];
    float* out = (float*)d_out;

    char* ws = (char*)d_ws;
    auto carve = [&](size_t bytes) -> char* {
        char* p = ws;
        ws += (bytes + 255) & ~(size_t)255;
        return p;
    };

    const int NT4H = (4 * H_) / 16;  // 256 N-tiles for the 4H gate dimension
    const int KC_I = I_ / 32;        // 16
    const int KC_H = H_ / 32;        // 32
    const int NT_O = O_ / 16;        // 32

    unsigned short* pWih1 = (unsigned short*)carve((size_t)NT4H * KC_I * 512 * 2);
    unsigned short* pWhh1 = (unsigned short*)carve((size_t)NT4H * KC_H * 512 * 2);
    unsigned short* pWih2 = (unsigned short*)carve((size_t)NT4H * KC_H * 512 * 2);
    unsigned short* pWhh2 = (unsigned short*)carve((size_t)NT4H * KC_H * 512 * 2);
    unsigned short* pFcW  = (unsigned short*)carve((size_t)NT_O * KC_H * 512 * 2);
    float* bias1 = (float*)carve((size_t)4 * H_ * 4);
    float* bias2 = (float*)carve((size_t)4 * H_ * 4);
    unsigned short* xb    = (unsigned short*)carve((size_t)B_ * T_ * I_ * 2);
    float* xproj          = (float*)carve((size_t)B_ * T_ * 4 * H_ * 4);     // 512 MB
    unsigned short* h1buf = (unsigned short*)carve((size_t)(T_ + 1) * B_ * H_ * 2);
    unsigned short* h2buf = (unsigned short*)carve((size_t)(T_ + 1) * B_ * H_ * 2);
    unsigned* bar1 = (unsigned*)carve(256);
    unsigned* bar2 = (unsigned*)carve(256);

    // --- pack weights into WMMA B-operand order (bf16) ---
    {
        int tot = NT4H * KC_I * 512;
        pack_b_kernel<<<(tot + 255) / 256, 256, 0, stream>>>(Wih1, pWih1, KC_I, I_, tot);
        tot = NT4H * KC_H * 512;
        pack_b_kernel<<<(tot + 255) / 256, 256, 0, stream>>>(Whh1, pWhh1, KC_H, H_, tot);
        pack_b_kernel<<<(tot + 255) / 256, 256, 0, stream>>>(Wih2, pWih2, KC_H, H_, tot);
        pack_b_kernel<<<(tot + 255) / 256, 256, 0, stream>>>(Whh2, pWhh2, KC_H, H_, tot);
        tot = NT_O * KC_H * 512;
        pack_b_kernel<<<(tot + 255) / 256, 256, 0, stream>>>(fcW, pFcW, KC_H, H_, tot);
    }
    vec_add_kernel<<<(4 * H_ + 255) / 256, 256, 0, stream>>>(bih1, bhh1, bias1, 4 * H_);
    vec_add_kernel<<<(4 * H_ + 255) / 256, 256, 0, stream>>>(bih2, bhh2, bias2, 4 * H_);
    {
        int tot = B_ * T_ * I_;
        cast_bf16_kernel<<<(tot + 255) / 256, 256, 0, stream>>>(x, xb, tot);
    }
    (void)hipMemsetAsync(h1buf, 0, (size_t)B_ * H_ * 2, stream);  // h0 = 0
    (void)hipMemsetAsync(h2buf, 0, (size_t)B_ * H_ * 2, stream);
    (void)hipMemsetAsync(bar1, 0, 256, stream);
    (void)hipMemsetAsync(bar2, 0, 256, stream);

    // --- layer 1: input projection (rows = b*T + t), then persistent recurrence ---
    {
        dim3 g(32768 / 128, 4096 / 64);
        gemm_bf16_wmma<<<g, 256, 0, stream>>>(xb, pWih1, bias1, xproj,
                                              32768, 4096, 512, 0, 0);
    }
    lstm_layer_persistent<<<32, 256, 0, stream>>>(
        h1buf, pWhh1, xproj,
        (long long)(4 * H_),           // xproj row for (b,t) = t*4H + b*T*4H
        (long long)T_ * 4 * H_,
        bar1, T_);

    // --- layer 2: input projection over h1 (rows = t*B + b), recurrence ---
    {
        dim3 g(32768 / 128, 4096 / 64);
        gemm_bf16_wmma<<<g, 256, 0, stream>>>(h1buf + (size_t)B_ * H_, pWih2, bias2,
                                              xproj, 32768, 4096, 1024, 0, 0);
    }
    lstm_layer_persistent<<<32, 256, 0, stream>>>(
        h2buf, pWhh2, xproj,
        (long long)B_ * 4 * H_,        // xproj row for (t,b) = t*B*4H + b*4H
        (long long)(4 * H_),
        bar2, T_);

    // --- final FC: h2 [t*B+b, H] -> out [b, t, O] (row remap in epilogue) ---
    {
        dim3 g(32768 / 128, 512 / 64);
        gemm_bf16_wmma<<<g, 256, 0, stream>>>(h2buf + (size_t)B_ * H_, pFcW, fcb,
                                              out, 32768, 512, 1024, B_, T_);
    }
}
